// TrajectoryAttention_5952824673115
// MI455X (gfx1250) — matrix operand order repaired and checked by reading, not compiled, not run on hardware
//
#include <hip/hip_runtime.h>

#define PP 920
#define FF 5
#define HH 8
#define CC 256
#define NN (FF * PP)   // 4600

typedef __attribute__((ext_vector_type(16))) __bf16 bf16x16;
typedef __attribute__((ext_vector_type(8)))  __bf16 bf16x8;
typedef __attribute__((ext_vector_type(8)))  float  f32x8;
typedef __attribute__((ext_vector_type(4)))  unsigned int u32x4;
typedef __attribute__((ext_vector_type(8)))  int i32x8;
typedef __attribute__((ext_vector_type(4)))  int i32x4;

__device__ inline f32x8 zero8f() {
    f32x8 r;
#pragma unroll
    for (int i = 0; i < 8; ++i) r[i] = 0.0f;
    return r;
}
__device__ inline bf16x8 zero8b() {
    bf16x8 r;
#pragma unroll
    for (int i = 0; i < 8; ++i) r[i] = (__bf16)0.0f;
    return r;
}
__device__ inline bf16x16 cat16(bf16x8 lo, bf16x8 hi) {
    bf16x16 r;
#pragma unroll
    for (int i = 0; i < 8; ++i) { r[i] = lo[i]; r[i + 8] = hi[i]; }
    return r;
}
__device__ inline bf16x8 ld8(const __bf16* p) { return *(const bf16x8*)p; }

// ----------------------------------------------------------------- TDM helper
// 2D tile load global->LDS via Tensor Data Mover (bf16 elements, data_size=2B).
// D# built per CDNA5 ISA 8.3/8.4:
//   group0: count=1 | lds_addr | global_addr[56:0] | type=2
//   group1: data_size=1(2B), tensor_dim0/1 (OOB rows/cols zero-filled),
//           tile_dim0/1, tensor_dim0_stride (elements)
__device__ inline void tdm_load_2d_bf16(unsigned lds_addr, const void* gptr,
                                        unsigned tensor_d0, unsigned tensor_d1,
                                        unsigned tile_d0, unsigned tile_d1,
                                        unsigned stride0) {
    unsigned long long ga = (unsigned long long)gptr;
    u32x4 g0;
    g0[0] = 1u;                                        // count = 1 (valid D#)
    g0[1] = lds_addr;                                  // LDS byte address
    g0[2] = (unsigned)(ga & 0xffffffffu);              // global addr [31:0]
    g0[3] = (unsigned)((ga >> 32) & 0x01ffffffu) | (2u << 30);  // addr hi | type=2
    i32x8 g1;
    g1[0] = (int)(1u << 16);                           // data_size = 2 bytes
    g1[1] = (int)((tensor_d0 & 0xffffu) << 16);        // tensor_dim0 lo16
    g1[2] = (int)((tensor_d0 >> 16) | ((tensor_d1 & 0xffffu) << 16));
    g1[3] = (int)((tensor_d1 >> 16) | (tile_d0 << 16));
    g1[4] = (int)(tile_d1 & 0xffffu);                  // tile_dim1 | tile_dim2=0
    g1[5] = (int)stride0;                              // dim0 stride lo32 (elems)
    g1[6] = 0;
    g1[7] = 0;
    i32x4 z4; z4[0] = 0; z4[1] = 0; z4[2] = 0; z4[3] = 0;
#if defined(__clang_major__) && (__clang_major__ >= 23)
    i32x8 z8;
#pragma unroll
    for (int i = 0; i < 8; ++i) z8[i] = 0;
    __builtin_amdgcn_tensor_load_to_lds(g0, g1, z4, z4, z8, 0);
#else
    __builtin_amdgcn_tensor_load_to_lds(g0, g1, z4, z4, 0);
#endif
}

__device__ inline unsigned lds_addr_of(const void* p) {
    return (unsigned)(unsigned long long)p;   // flat LDS aperture: addr[31:0] = LDS offset
}

// ---------------------------------------------------------------- cvt f32->bf16
__global__ void cvt_f32_bf16(const float* __restrict__ src, __bf16* __restrict__ dst, int n) {
    int i = blockIdx.x * blockDim.x + threadIdx.x;
    if (i < n) dst[i] = (__bf16)src[i];
}

// ---------------------------------------------------------- permute x -> xb bf16
// x: (P, B*F, C) f32  ->  xb: (B, N=F*P, C) bf16
__global__ void pack_x(const float* __restrict__ x, __bf16* __restrict__ xb, int B) {
    size_t idx = (size_t)blockIdx.x * blockDim.x + threadIdx.x;
    size_t total = (size_t)B * NN * CC;
    if (idx >= total) return;
    int c = (int)(idx % CC);
    size_t r = idx / CC;
    int s = (int)(r % NN);
    int b = (int)(r / NN);
    int f = s / PP, p = s % PP;
    size_t src = ((size_t)p * (B * FF) + (size_t)b * FF + f) * CC + c;
    xb[idx] = (__bf16)x[src];
}

// ------------------------------------------------------------------ WMMA GEMM
// C[m,n] = sum_k A[m,k] * W[n,k]   (K = 256 fixed; A,W bf16 row-major)
// mode 0: store bf16 row-major (ld = Nout)
// mode 1: store f32 to d_out with (b,s)->(p, b*F+f) row permutation + bias
__global__ __launch_bounds__(128) void gemm_bf16(
    const __bf16* __restrict__ A, const __bf16* __restrict__ W,
    int M, int Nout,
    __bf16* __restrict__ outb, float* __restrict__ outf,
    const float* __restrict__ bias, int mode, int B)
{
    __shared__ __bf16 At[64 * 32];
    __shared__ __bf16 Wt[64 * 32];
    const int t = threadIdx.x;
    const int lane = t & 31;
    const int wave = t >> 5;
    const int half = lane >> 4;
    const int l15  = lane & 15;
    const int m0 = blockIdx.x * 64;
    const int n0 = blockIdx.y * 64;

    f32x8 acc[4];
#pragma unroll
    for (int i = 0; i < 4; ++i) acc[i] = zero8f();

    for (int kb = 0; kb < 8; ++kb) {
#pragma unroll
        for (int i = 0; i < 2; ++i) {
            int chunk = t * 2 + i;            // 0..255
            int rowi = chunk >> 2;
            int col8 = (chunk & 3) * 8;
            int ga = m0 + rowi;
            bf16x8 va = zero8b();
            if (ga < M) va = ld8(A + (size_t)ga * 256 + kb * 32 + col8);
            *(bf16x8*)(At + rowi * 32 + col8) = va;
            int gw = n0 + rowi;
            bf16x8 vw = ld8(W + (size_t)gw * 256 + kb * 32 + col8);
            *(bf16x8*)(Wt + rowi * 32 + col8) = vw;
        }
        __syncthreads();
        const __bf16* ar = At + (wave * 16 + l15) * 32;
        bf16x16 af = cat16(ld8(ar + 16 * half), ld8(ar + 16 * half + 8));
#pragma unroll
        for (int nt = 0; nt < 4; ++nt) {
            const __bf16* br = Wt + (nt * 16 + l15) * 32 + 16 * half;
            bf16x16 bfv = cat16(ld8(br), ld8(br + 8));
            acc[nt] = __builtin_amdgcn_wmma_f32_16x16x32_bf16(
                false, af, false, bfv, (short)0, acc[nt], false, false);
        }
        __syncthreads();
    }

#pragma unroll
    for (int nt = 0; nt < 4; ++nt) {
#pragma unroll
        for (int j = 0; j < 8; ++j) {
            int row = m0 + wave * 16 + j + 8 * half;
            int col = n0 + nt * 16 + l15;
            if (row >= M) continue;
            float v = acc[nt][j];
            if (mode == 0) {
                outb[(size_t)row * Nout + col] = (__bf16)v;
            } else {
                v += bias[col];
                int b = row / NN;
                int s = row % NN;
                int f = s / PP, p = s % PP;
                outf[((size_t)p * (B * FF) + (size_t)b * FF + f) * CC + col] = v;
            }
        }
    }
}

// ------------------------------------------------ stage-1 flash space attention
// qkv: (B*N, 768) bf16 rows = q|k|v.  One wave handles a 16-query tile for (b,f),
// streaming 920 keys of frame f in 32-key chunks with online softmax.
// Q/K tiles are staged into LDS by the Tensor Data Mover (TENSORcnt, OOB rows
// zero-filled by hardware); V is staged transposed by the wave, overlapped with
// the in-flight TDM transfer.
// Writes x1: (B*N*F, 256) bf16 and xdiag: (B*N, 256) bf16 (rows where f == s/P).
__global__ __launch_bounds__(32) void flash_space(
    const __bf16* __restrict__ qkv,
    __bf16* __restrict__ x1,
    __bf16* __restrict__ xdiag,
    int B)
{
    __shared__ __bf16 Qt[16 * 256];
    __shared__ __bf16 Kt[32 * 256];
    __shared__ __bf16 Vt[256 * 32];   // transposed: [channel][key]
    __shared__ __bf16 Pt[16 * 32];

    const int lane = threadIdx.x;
    const int half = lane >> 4;
    const int l15  = lane & 15;
    const int qt = blockIdx.x;        // 0..287
    const int f  = blockIdx.y;        // 0..4
    const int b  = blockIdx.z;

    const float scale = 0.17677669529663687f;   // 32^-0.5
    const size_t baseRow = (size_t)b * NN;
    const unsigned qt_lds = lds_addr_of(Qt);
    const unsigned kt_lds = lds_addr_of(Kt);

    // Q tile (16 x 256) via TDM; rows past N are zero-filled by tensor_dim1.
    tdm_load_2d_bf16(qt_lds, qkv + (baseRow + (size_t)qt * 16) * 768,
                     256u, (unsigned)(NN - qt * 16), 256u, 16u, 768u);
    __builtin_amdgcn_s_wait_tensorcnt(0);
    __syncthreads();

    bf16x16 aq[8];
#pragma unroll
    for (int ks = 0; ks < 8; ++ks) {
        const __bf16* ar = Qt + l15 * 256 + ks * 32;
        aq[ks] = cat16(ld8(ar + 16 * half), ld8(ar + 16 * half + 8));
    }

    f32x8 o[16];
#pragma unroll
    for (int i = 0; i < 16; ++i) o[i] = zero8f();
    float mrow[8], lrow[8];
#pragma unroll
    for (int j = 0; j < 8; ++j) { mrow[j] = -1e30f; lrow[j] = 0.0f; }

    for (int kc = 0; kc < 29; ++kc) {          // 29*32 = 928 >= 920
        __syncthreads();
        // K tile (32 x 256) via TDM; tail rows (>=920) zero-filled in hardware.
        tdm_load_2d_bf16(kt_lds,
                         qkv + (baseRow + (size_t)f * PP + (size_t)kc * 32) * 768 + 256,
                         256u, (unsigned)(PP - kc * 32), 256u, 32u, 768u);
        // V tile staged transposed by the wave, overlapping the TDM transfer.
        for (int ch = lane; ch < 32 * 32; ch += 32) {
            int r = ch >> 5;
            int c8 = (ch & 31) * 8;
            int kk = kc * 32 + r;
            bf16x8 vv = zero8b();
            if (kk < PP)
                vv = ld8(qkv + (baseRow + (size_t)f * PP + kk) * 768 + 512 + c8);
#pragma unroll
            for (int i = 0; i < 8; ++i) Vt[(c8 + i) * 32 + r] = vv[i];
        }
        __builtin_amdgcn_s_wait_tensorcnt(0);
        __syncthreads();

        f32x8 s0 = zero8f(), s1 = zero8f();
#pragma unroll
        for (int ks = 0; ks < 8; ++ks) {
            const __bf16* b0 = Kt + l15 * 256 + ks * 32 + 16 * half;
            bf16x16 bf0 = cat16(ld8(b0), ld8(b0 + 8));
            s0 = __builtin_amdgcn_wmma_f32_16x16x32_bf16(
                false, aq[ks], false, bf0, (short)0, s0, false, false);
            const __bf16* b1 = Kt + (16 + l15) * 256 + ks * 32 + 16 * half;
            bf16x16 bf1 = cat16(ld8(b1), ld8(b1 + 8));
            s1 = __builtin_amdgcn_wmma_f32_16x16x32_bf16(
                false, aq[ks], false, bf1, (short)0, s1, false, false);
        }

        const int kk0 = kc * 32 + l15;
        const int kk1 = kc * 32 + 16 + l15;
        float p0[8], p1[8], alpha[8];
#pragma unroll
        for (int j = 0; j < 8; ++j) {
            float v0 = (kk0 < PP) ? s0[j] * scale : -1e30f;
            float v1 = (kk1 < PP) ? s1[j] * scale : -1e30f;
            float mx = fmaxf(v0, v1);
#pragma unroll
            for (int off = 1; off < 16; off <<= 1) mx = fmaxf(mx, __shfl_xor(mx, off, 16));
            float mnew = fmaxf(mrow[j], mx);
            alpha[j] = __expf(mrow[j] - mnew);
            mrow[j] = mnew;
            p0[j] = __expf(v0 - mnew);
            p1[j] = __expf(v1 - mnew);
            float rs = p0[j] + p1[j];
#pragma unroll
            for (int off = 1; off < 16; off <<= 1) rs += __shfl_xor(rs, off, 16);
            lrow[j] = alpha[j] * lrow[j] + rs;
        }
#pragma unroll
        for (int ct = 0; ct < 16; ++ct)
#pragma unroll
            for (int j = 0; j < 8; ++j) o[ct][j] *= alpha[j];

#pragma unroll
        for (int j = 0; j < 8; ++j) {
            Pt[(j + 8 * half) * 32 + l15]      = (__bf16)p0[j];
            Pt[(j + 8 * half) * 32 + 16 + l15] = (__bf16)p1[j];
        }
        __syncthreads();
        const __bf16* pr = Pt + l15 * 32;
        bf16x16 ap = cat16(ld8(pr + 16 * half), ld8(pr + 16 * half + 8));
#pragma unroll
        for (int ct = 0; ct < 16; ++ct) {
            const __bf16* vr = Vt + (ct * 16 + l15) * 32 + 16 * half;
            bf16x16 bv = cat16(ld8(vr), ld8(vr + 8));
            o[ct] = __builtin_amdgcn_wmma_f32_16x16x32_bf16(
                false, ap, false, bv, (short)0, o[ct], false, false);
        }
    }

    float linv[8];
#pragma unroll
    for (int j = 0; j < 8; ++j) linv[j] = (lrow[j] > 0.0f) ? 1.0f / lrow[j] : 0.0f;

#pragma unroll
    for (int ct = 0; ct < 16; ++ct) {
#pragma unroll
        for (int j = 0; j < 8; ++j) {
            int q = qt * 16 + j + 8 * half;
            if (q >= NN) continue;
            int c = ct * 16 + l15;
            __bf16 hv = (__bf16)(o[ct][j] * linv[j]);
            size_t row = baseRow + q;
            x1[(row * FF + f) * 256 + c] = hv;
            if (q / PP == f) xdiag[row * 256 + c] = hv;
        }
    }
}

// ------------------------------------- stage-2 trajectory attention (tiny, d=32)
__global__ void traj_attn(const __bf16* __restrict__ q2,
                          const __bf16* __restrict__ kv2,
                          __bf16* __restrict__ x2,
                          float* __restrict__ attn_out,
                          int B)
{
    int idx = blockIdx.x * blockDim.x + threadIdx.x;
    int total = B * NN * HH;
    if (idx >= total) return;
    int h = idx % HH;
    int rest = idx / HH;
    int s = rest % NN;
    int b = rest / NN;
    const float scale = 0.17677669529663687f;
    size_t row = (size_t)b * NN + s;

    const __bf16* qp = q2 + row * 256 + h * 32;
    float qv[32];
#pragma unroll
    for (int i = 0; i < 32; ++i) qv[i] = (float)qp[i] * scale;

    float logit[5];
#pragma unroll
    for (int f = 0; f < 5; ++f) {
        const __bf16* kp = kv2 + (row * FF + f) * 512 + h * 32;
        float acc = 0.0f;
#pragma unroll
        for (int i = 0; i < 32; ++i) acc += qv[i] * (float)kp[i];
        logit[f] = acc;
    }
    float mx = logit[0];
#pragma unroll
    for (int f = 1; f < 5; ++f) mx = fmaxf(mx, logit[f]);
    float w[5], den = 0.0f;
#pragma unroll
    for (int f = 0; f < 5; ++f) { w[f] = __expf(logit[f] - mx); den += w[f]; }
    float dinv = 1.0f / den;
#pragma unroll
    for (int f = 0; f < 5; ++f) {
        w[f] *= dinv;
        attn_out[(((size_t)b * HH + h) * NN + s) * FF + f] = w[f];
    }
    float ov[32];
#pragma unroll
    for (int i = 0; i < 32; ++i) ov[i] = 0.0f;
#pragma unroll
    for (int f = 0; f < 5; ++f) {
        const __bf16* vp = kv2 + (row * FF + f) * 512 + 256 + h * 32;
#pragma unroll
        for (int i = 0; i < 32; ++i) ov[i] += w[f] * (float)vp[i];
    }
    __bf16* xp = x2 + row * 256 + h * 32;
#pragma unroll
    for (int i = 0; i < 32; ++i) xp[i] = (__bf16)ov[i];
}

// --------------------------------------------------------------------- launch
extern "C" void kernel_launch(void* const* d_in, const int* in_sizes, int n_in,
                              void* d_out, int out_size, void* d_ws, size_t ws_size,
                              hipStream_t stream)
{
    (void)n_in; (void)out_size; (void)ws_size;
    const float* x      = (const float*)d_in[0];
    const float* W_qkv  = (const float*)d_in[1];
    const float* W_pq   = (const float*)d_in[2];
    const float* W_pkv  = (const float*)d_in[3];
    const float* W_proj = (const float*)d_in[4];
    const float* b_proj = (const float*)d_in[5];

    const int BF = in_sizes[0] / (PP * CC);   // B*F = 20
    const int B  = BF / FF;                   // 4
    const size_t BN = (size_t)B * NN;

    __bf16* ws = (__bf16*)d_ws;
    size_t off = 0;
    __bf16* xb    = ws + off; off += BN * CC;
    __bf16* wq    = ws + off; off += (size_t)768 * 256;
    __bf16* wpq   = ws + off; off += (size_t)256 * 256;
    __bf16* wpkv  = ws + off; off += (size_t)512 * 256;
    __bf16* wproj = ws + off; off += (size_t)256 * 256;
    __bf16* qkv   = ws + off; off += BN * 768;
    __bf16* x1    = ws + off; off += BN * FF * CC;
    __bf16* xdiag = ws + off; off += BN * CC;
    __bf16* q2    = ws + off; off += BN * CC;
    __bf16* kv2   = ws + off; off += BN * FF * 512;
    __bf16* x2    = ws + off; off += BN * CC;

    float* outp  = (float*)d_out;
    float* attnp = outp + (size_t)PP * BF * CC;

    cvt_f32_bf16<<<(768 * 256 + 255) / 256, 256, 0, stream>>>(W_qkv, wq, 768 * 256);
    cvt_f32_bf16<<<(256 * 256 + 255) / 256, 256, 0, stream>>>(W_pq, wpq, 256 * 256);
    cvt_f32_bf16<<<(512 * 256 + 255) / 256, 256, 0, stream>>>(W_pkv, wpkv, 512 * 256);
    cvt_f32_bf16<<<(256 * 256 + 255) / 256, 256, 0, stream>>>(W_proj, wproj, 256 * 256);

    {
        size_t tot = BN * CC;
        pack_x<<<(unsigned)((tot + 255) / 256), 256, 0, stream>>>(x, xb, B);
    }
    {
        int M = (int)BN;
        dim3 g((M + 63) / 64, 768 / 64);
        gemm_bf16<<<g, 128, 0, stream>>>(xb, wq, M, 768, qkv, nullptr, nullptr, 0, B);
    }
    {
        dim3 g((NN + 15) / 16, FF, B);
        flash_space<<<g, 32, 0, stream>>>(qkv, x1, xdiag, B);
    }
    {
        int M = (int)(BN * FF);
        dim3 g((M + 63) / 64, 512 / 64);
        gemm_bf16<<<g, 128, 0, stream>>>(x1, wpkv, M, 512, kv2, nullptr, nullptr, 0, B);
    }
    {
        int M = (int)BN;
        dim3 g((M + 63) / 64, 256 / 64);
        gemm_bf16<<<g, 128, 0, stream>>>(xdiag, wpq, M, 256, q2, nullptr, nullptr, 0, B);
    }
    {
        int total = B * NN * HH;
        traj_attn<<<(total + 255) / 256, 256, 0, stream>>>(q2, kv2, x2, attnp, B);
    }
    {
        int M = (int)BN;
        dim3 g((M + 63) / 64, 256 / 64);
        gemm_bf16<<<g, 128, 0, stream>>>(x2, wproj, M, 256, nullptr, outp, b_proj, 1, B);
    }
}
